// FlaxFullyConnectedTensorProduct_60198261620948
// MI455X (gfx1250) — matrix operand order repaired and checked by reading, not compiled, not run on hardware
//
#include <hip/hip_runtime.h>

typedef __bf16 v16bf __attribute__((ext_vector_type(16)));
typedef float  v8f   __attribute__((ext_vector_type(8)));

#define WMMA_BF16(A, Bm, C) \
    __builtin_amdgcn_wmma_f32_16x16x32_bf16(false, (A), false, (Bm), (short)0, (C), false, false)

// out column layout: [0..15]=out0, [16..63]=out1e (3w+k), [64..111]=out1o (3w+k)
__global__ __launch_bounds__(256) void ftp_kernel(
    const float* __restrict__ x1,  const float* __restrict__ x2,
    const float* __restrict__ w000, const float* __restrict__ w110,
    const float* __restrict__ w011, const float* __restrict__ w101,
    const float* __restrict__ w111, float* __restrict__ out)
{
    // path normalisation constants folded into the weights
    const float S000 = sqrtf(1.0f / 512.0f);                 // A0
    const float S110 = sqrtf(1.0f / 512.0f) / sqrtf(3.0f);   // A0/sqrt(3)
    const float S111 = sqrtf(3.0f / 256.0f) / sqrtf(6.0f);   // A1E/sqrt(6)
    const float S1O  = sqrtf(3.0f / 512.0f) / sqrtf(3.0f);   // A1O/sqrt(3)

    // Stage bf16 weights in LDS, layout [u/2][v/8][w][u&1][v&7] so a lane's B-fragment
    // (u = 2c, 2c+1, fixed output column w, v = 8h..8h+7) is 32 contiguous bytes.
    __shared__ __align__(32) __bf16 Lw[5][4096];
    const int tid = threadIdx.x;
    {
        const float* wsrc[5] = { w000, w110, w111, w011, w101 };
        const float  wscl[5] = { S000, S110, S111, S1O,  S1O  };
        #pragma unroll
        for (int m = 0; m < 5; ++m) {
            const float* ws = wsrc[m];
            const float  sc = wscl[m];
            for (int idx = tid; idx < 4096; idx += 256) {
                int u = idx >> 8, v = (idx >> 4) & 15, n = idx & 15; // src: [u][v][n]
                Lw[m][(((u & ~1) | (v >> 3)) * 16 + n) * 16 + (((u & 1) << 3) | (v & 7))] = (__bf16)(sc * ws[idx]);
            }
        }
    }
    __syncthreads();

    const int lane = tid & 31;
    const int wv   = tid >> 5;
    const int half = lane >> 4;       // 0: K 0..7/16..23 (A & B); 1: other halves
    const int zl   = lane & 15;       // A-matrix row within tile
    const int n    = lane & 15;       // B/C/D column
    const int zbase = blockIdx.x * 128 + wv * 16;
    const int z     = zbase + zl;

    // Per-lane operand data (z = lane&15; duplicated across lane halves for x1,
    // split by v-half for x2).
    float s1[16], v1[16][3], s2[8], v2[8][3];
    {
        const float* r1 = x1 + (size_t)z * 64;
        #pragma unroll
        for (int u = 0; u < 16; ++u) s1[u] = r1[u];
        #pragma unroll
        for (int u = 0; u < 16; ++u)
            #pragma unroll
            for (int i = 0; i < 3; ++i) v1[u][i] = r1[16 + 3 * u + i];
        const float* r2 = x2 + (size_t)z * 64;
        const int vlo = half * 8;
        #pragma unroll
        for (int j = 0; j < 8; ++j) s2[j] = r2[vlo + j];
        #pragma unroll
        for (int j = 0; j < 8; ++j)
            #pragma unroll
            for (int i = 0; i < 3; ++i) v2[j][i] = r2[16 + 3 * (vlo + j) + i];
    }

    v8f acc0 = {};
    v8f acc1e[3] = {};
    v8f acc1o[3] = {};

    // K = 256 over flattened (u,v); chunk c covers u ∈ {2c, 2c+1}, v ∈ 0..15.
    #pragma unroll
    for (int c = 0; c < 8; ++c) {
        const int u0 = 2 * c, u1 = 2 * c + 1;
        const int ub = 2 * c + half; // LDS row 2c + v-half; B fragment spans u0, u1

        const v16bf b000 = *(const v16bf*)&Lw[0][(ub * 16 + n) * 16];
        const v16bf b110 = *(const v16bf*)&Lw[1][(ub * 16 + n) * 16];
        const v16bf b111 = *(const v16bf*)&Lw[2][(ub * 16 + n) * 16];
        const v16bf b011 = *(const v16bf*)&Lw[3][(ub * 16 + n) * 16];
        const v16bf b101 = *(const v16bf*)&Lw[4][(ub * 16 + n) * 16];

        v16bf a000, a110, aX[3], aS1V2[3], aV1S2[3];
        #pragma unroll
        for (int j = 0; j < 8; ++j) {
            // element j      <-> (u0, vlo+j)
            // element 8 + j  <-> (u1, vlo+j)
            a000[j]     = (__bf16)(s1[u0] * s2[j]);
            a000[8 + j] = (__bf16)(s1[u1] * s2[j]);

            a110[j]     = (__bf16)(v1[u0][0] * v2[j][0] + v1[u0][1] * v2[j][1] + v1[u0][2] * v2[j][2]);
            a110[8 + j] = (__bf16)(v1[u1][0] * v2[j][0] + v1[u1][1] * v2[j][1] + v1[u1][2] * v2[j][2]);

            aX[0][j]     = (__bf16)(v1[u0][1] * v2[j][2] - v1[u0][2] * v2[j][1]);
            aX[0][8 + j] = (__bf16)(v1[u1][1] * v2[j][2] - v1[u1][2] * v2[j][1]);
            aX[1][j]     = (__bf16)(v1[u0][2] * v2[j][0] - v1[u0][0] * v2[j][2]);
            aX[1][8 + j] = (__bf16)(v1[u1][2] * v2[j][0] - v1[u1][0] * v2[j][2]);
            aX[2][j]     = (__bf16)(v1[u0][0] * v2[j][1] - v1[u0][1] * v2[j][0]);
            aX[2][8 + j] = (__bf16)(v1[u1][0] * v2[j][1] - v1[u1][1] * v2[j][0]);

            #pragma unroll
            for (int k = 0; k < 3; ++k) {
                aS1V2[k][j]     = (__bf16)(s1[u0] * v2[j][k]);
                aS1V2[k][8 + j] = (__bf16)(s1[u1] * v2[j][k]);
                aV1S2[k][j]     = (__bf16)(v1[u0][k] * s2[j]);
                aV1S2[k][8 + j] = (__bf16)(v1[u1][k] * s2[j]);
            }
        }

        acc0 = WMMA_BF16(a000, b000, acc0);
        acc0 = WMMA_BF16(a110, b110, acc0);
        #pragma unroll
        for (int k = 0; k < 3; ++k) {
            acc1e[k] = WMMA_BF16(aX[k],    b111, acc1e[k]);
            acc1o[k] = WMMA_BF16(aS1V2[k], b011, acc1o[k]);
            acc1o[k] = WMMA_BF16(aV1S2[k], b101, acc1o[k]);
        }
    }

    // C/D layout: lane holds column n = lane&15, rows M = r + 8*half (r = VGPR idx)
    const int rowM = half * 8;
    #pragma unroll
    for (int r = 0; r < 8; ++r) {
        float* row = out + (size_t)(zbase + rowM + r) * 112;
        row[n] = acc0[r];
        #pragma unroll
        for (int k = 0; k < 3; ++k) {
            row[16 + 3 * n + k] = acc1e[k][r];
            row[64 + 3 * n + k] = acc1o[k][r];
        }
    }
}

extern "C" void kernel_launch(void* const* d_in, const int* in_sizes, int n_in,
                              void* d_out, int out_size, void* d_ws, size_t ws_size,
                              hipStream_t stream) {
    const float* x1   = (const float*)d_in[0];
    const float* x2   = (const float*)d_in[1];
    const float* w000 = (const float*)d_in[2];
    const float* w110 = (const float*)d_in[3];
    const float* w011 = (const float*)d_in[4];
    const float* w101 = (const float*)d_in[5];
    const float* w111 = (const float*)d_in[6];
    float* out = (float*)d_out;

    const int Bn = in_sizes[0] / 64;     // 16384 batch rows
    const int blocks = Bn / 128;         // 128 rows (8 waves x 16) per block
    ftp_kernel<<<blocks, 256, 0, stream>>>(x1, x2, w000, w110, w011, w101, w111, out);
}
